// CrossTeamInteraction_1391569404143
// MI455X (gfx1250) — matrix operand order repaired and checked by reading, not compiled, not run on hardware
//
#include <hip/hip_runtime.h>
#include <hip/hip_bf16.h>
#include <math.h>

// ---------------------------------------------------------------------------
// CrossTeamInteraction fused kernel for MI455X (gfx1250, wave32, WMMA).
//   B=32768 batches, L=11, D=32, H=4 heads, HD=8, PE=16.
// One wave32 per batch item; all dense projections via
// v_wmma_f32_16x16x32_f16 (K=32 == D, one WMMA per 16x16 output tile).
// K/V kept f32 in LDS to avoid per-MAC cvt in the VALU softmax loops.
// ---------------------------------------------------------------------------

typedef __attribute__((ext_vector_type(16))) _Float16 v16h;
typedef __attribute__((ext_vector_type(8)))  float    v8f;

#define LL   11
#define DD   32
#define PE   16
#define NW   4      // waves per block (128 threads)

// CDNA5 wave-local LDS fence: lanes are lockstep, so once the wave's DS ops
// retire (DScnt==0) every lane sees every lane's LDS writes.
__device__ __forceinline__ void wave_lds_sync() {
  asm volatile("s_wait_dscnt 0" ::: "memory");
}

__device__ __forceinline__ v8f wmma_f16(v16h a, v16h b, v8f c) {
  // (neg_a, A, neg_b, B, c_mod, C, reuse_a, reuse_b)
  return __builtin_amdgcn_wmma_f32_16x16x32_f16(false, a, false, b,
                                                (short)0, c, false, false);
}

// A-matrix 16x32 f16 fragment (ISA 7.12.2):
//   lanes 0-15 : row M=lane,    K=0..7 (v0-3) and K=16..23 (v4-7)
//   lanes16-31 : row M=lane-16, K=8..15        and K=24..31
__device__ __forceinline__ v16h load_a_frag(const _Float16* x, int lane) {
  const int m  = lane & 15;
  const int k0 = (lane & 16) ? 8 : 0;
  const _Float16* p = x + m * DD + k0;
  v16h f;
#pragma unroll
  for (int i = 0; i < 8; ++i) { f[i] = p[i]; f[8 + i] = p[16 + i]; }
  return f;
}

// B-matrix 32x16 f16 fragment where B[k][n] = W[row0+n][k]
// (W row-major [32 cols], i.e. x @ W^T):
//   lanes 0-15 : col N=lane,    K=0..7 and K=16..23 ; lanes16-31: N=lane-16, K=8..15 and K=24..31
__device__ __forceinline__ v16h load_b_frag(const _Float16* w, int row0, int lane) {
  const _Float16* p = w + (row0 + (lane & 15)) * DD + ((lane >> 4) << 3);
  v16h f;
#pragma unroll
  for (int i = 0; i < 8; ++i) { f[i] = p[i]; f[8 + i] = p[16 + i]; }
  return f;
}

__global__ __launch_bounds__(128, 1)
void crossteam_kernel(
    const int* __restrict__ home_p,  const int* __restrict__ away_p,
    const int* __restrict__ h_pos,   const int* __restrict__ a_pos,
    const float* __restrict__ emb,   const float* __restrict__ pos_emb,
    const float* __restrict__ h2a_in_w, const float* __restrict__ h2a_in_b,
    const float* __restrict__ h2a_out_w, const float* __restrict__ h2a_out_b,
    const float* __restrict__ a2h_in_w, const float* __restrict__ a2h_in_b,
    const float* __restrict__ a2h_out_w, const float* __restrict__ a2h_out_b,
    const float* __restrict__ gate_w, const float* __restrict__ gate_b,
    const float* __restrict__ ln_w,   const float* __restrict__ ln_b,
    float* __restrict__ out, int nbatch)
{
  // ---- block-shared weights (converted once, reused by all batches)
  __shared__ _Float16 s_w_in [2][3 * DD * DD];   // [dir][Wq;Wk;Wv] 96x32
  __shared__ _Float16 s_w_out[2][DD * DD];       // 32x32
  __shared__ float    s_b_in [2][3 * DD];
  __shared__ float    s_b_out[2][DD];
  __shared__ float    s_gate_w[DD * 2 * DD];     // 32x64
  __shared__ float    s_gate_b[DD], s_ln_w[DD], s_ln_b[DD];
  // ---- per-wave working tiles
  __shared__ _Float16 s_x [NW][2][16 * DD];      // padded 16x32 f16 input tiles
  __shared__ float    s_q [NW][16 * DD];         // pre-scaled by 1/sqrt(HD)
  __shared__ float    s_k [NW][16 * DD];         // f32: no cvt in score loop
  __shared__ float    s_v [NW][16 * DD];         // f32: no cvt in AV loop
  __shared__ _Float16 s_o [NW][16 * DD];         // attn out (A-frag for outproj)
  __shared__ float    s_vec [NW][2 * DD];        // [h2a_vec ; a2h_vec]
  __shared__ float    s_part[NW][2 * DD];

  const int tid  = threadIdx.x;
  const int lane = tid & 31;
  const int wv   = tid >> 5;

  // stage weights -> LDS (f16 for WMMA operands, f32 for bias/gate/ln)
  for (int i = tid; i < 3 * DD * DD; i += blockDim.x) {
    s_w_in[0][i] = (_Float16)h2a_in_w[i];
    s_w_in[1][i] = (_Float16)a2h_in_w[i];
  }
  for (int i = tid; i < DD * DD; i += blockDim.x) {
    s_w_out[0][i] = (_Float16)h2a_out_w[i];
    s_w_out[1][i] = (_Float16)a2h_out_w[i];
  }
  for (int i = tid; i < 3 * DD; i += blockDim.x) {
    s_b_in[0][i] = h2a_in_b[i];
    s_b_in[1][i] = a2h_in_b[i];
  }
  for (int i = tid; i < DD; i += blockDim.x) {
    s_b_out[0][i] = h2a_out_b[i];
    s_b_out[1][i] = a2h_out_b[i];
    s_gate_b[i] = gate_b[i]; s_ln_w[i] = ln_w[i]; s_ln_b[i] = ln_b[i];
  }
  for (int i = tid; i < DD * 2 * DD; i += blockDim.x) s_gate_w[i] = gate_w[i];
  __syncthreads();

  const int gw     = blockIdx.x * NW + wv;
  const int nwaves = gridDim.x * NW;
  const int mrow   = (lane >> 4) * 8;  // C-tile: row = mrow + j, col = ncol
  const int ncol   = lane & 15;
  const float inv_sqrt_hd = 0.35355339059327373f;

  for (int b = gw; b < nbatch; b += nwaves) {
    // prefetch next iteration's index rows while we work on this one
    if (b + nwaves < nbatch && lane == 0) {
      __builtin_prefetch(&home_p[(size_t)(b + nwaves) * LL], 0, 0);
      __builtin_prefetch(&away_p[(size_t)(b + nwaves) * LL], 0, 0);
      __builtin_prefetch(&h_pos [(size_t)(b + nwaves) * LL], 0, 0);
      __builtin_prefetch(&a_pos [(size_t)(b + nwaves) * LL], 0, 0);
    }

    // ---- 1. indices + padding masks (11 bits) -----------------------------
    int hp = 0, ap = 0, hq = 0, aq = 0;
    if (lane < LL) {
      hp = home_p[(size_t)b * LL + lane];
      ap = away_p[(size_t)b * LL + lane];
      hq = h_pos [(size_t)b * LL + lane];
      aq = a_pos [(size_t)b * LL + lane];
    }
    const unsigned hpad = (unsigned)__ballot(lane < LL && hp == 0) & 0x7FFu;
    const unsigned apad = (unsigned)__ballot(lane < LL && ap == 0) & 0x7FFu;
    // safe key-padding masks (all-pad row -> no masking)
    const unsigned hk = (hpad == 0x7FFu) ? 0u : hpad;
    const unsigned ak = (apad == 0x7FFu) ? 0u : apad;
    // query-aggregation valid-row masks (all-pad -> all rows valid)
    const unsigned hv = (hpad == 0x7FFu) ? 0x7FFu : (~hpad & 0x7FFu);
    const unsigned av = (apad == 0x7FFu) ? 0x7FFu : (~apad & 0x7FFu);

    // ---- 2. gather embeddings -> padded 16x32 f16 tiles -------------------
    // 32 lanes = 16 rows x 2 halves; row indices broadcast via wave shuffles.
    {
      const int row = lane >> 1, half = lane & 1;
      const int ip0 = __shfl(half ? hq : hp, row, 32);  // home idx for this row
      const int ip1 = __shfl(half ? aq : ap, row, 32);  // away idx
      const int rowsel[2] = {ip0, ip1};
#pragma unroll
      for (int t = 0; t < 2; ++t) {
        _Float16 tmp[16];
        if (row < LL) {
          const float* src = (half == 0 ? emb : pos_emb) + (size_t)rowsel[t] * PE;
#pragma unroll
          for (int i = 0; i < 16; ++i) tmp[i] = (_Float16)src[i];
        } else {
#pragma unroll
          for (int i = 0; i < 16; ++i) tmp[i] = (_Float16)0.f;
        }
        _Float16* dst = &s_x[wv][t][row * DD + half * 16];
#pragma unroll
        for (int i = 0; i < 16; ++i) dst[i] = tmp[i];
      }
    }
    wave_lds_sync();

    // ---- 3. two cross-attention directions --------------------------------
#pragma unroll
    for (int dir = 0; dir < 2; ++dir) {
      const _Float16* xq    = s_x[wv][dir];       // dir0: home queries -> away kv
      const _Float16* xkv   = s_x[wv][1 - dir];
      const unsigned  kmask = (dir == 0) ? ak : hk;
      const unsigned  qval  = (dir == 0) ? hv : av;
      const _Float16* w_in  = s_w_in[dir];
      const float*    b_in  = s_b_in[dir];
      const _Float16* w_out = s_w_out[dir];
      const float*    b_out = s_b_out[dir];

      const v16h a_q  = load_a_frag(xq,  lane);
      const v16h a_kv = load_a_frag(xkv, lane);

      // QKV projections: 6 x v_wmma_f32_16x16x32_f16 (one per 16x16 tile)
#pragma unroll
      for (int nt = 0; nt < 2; ++nt) {
        const int nb = nt * 16;
        v8f cq = {}; cq = wmma_f16(a_q,  load_b_frag(w_in,            nb, lane), cq);
        v8f ck = {}; ck = wmma_f16(a_kv, load_b_frag(w_in + DD * DD,  nb, lane), ck);
        v8f cv = {}; cv = wmma_f16(a_kv, load_b_frag(w_in + 2*DD*DD,  nb, lane), cv);
        const float bq = b_in[nb + ncol];
        const float bk = b_in[DD + nb + ncol];
        const float bv = b_in[2 * DD + nb + ncol];
#pragma unroll
        for (int j = 0; j < 8; ++j) {
          const int m = mrow + j;
          s_q[wv][m * DD + nb + ncol] = (cq[j] + bq) * inv_sqrt_hd; // fold scale
          s_k[wv][m * DD + nb + ncol] = ck[j] + bk;
          s_v[wv][m * DD + nb + ncol] = cv[j] + bv;
        }
      }
      wave_lds_sync();

      // zero attention-output tile (rows 11..15 must stay 0 for out-proj WMMA)
#pragma unroll
      for (int i = 0; i < 16; ++i) s_o[wv][lane * 16 + i] = (_Float16)0.f;
      wave_lds_sync();

      // softmax attention on VALU: 44 (head,query) pairs over 32 lanes,
      // all LDS traffic as float4 (ds_load_b128)
      for (int p = lane; p < 4 * LL; p += 32) {
        const int h = p / LL, i = p - h * LL;
        float4 q0 = *(const float4*)&s_q[wv][i * DD + h * 8];
        float4 q1 = *(const float4*)&s_q[wv][i * DD + h * 8 + 4];
        float sc[LL], mx = -1e30f;
#pragma unroll
        for (int j = 0; j < LL; ++j) {
          const float4 k0 = *(const float4*)&s_k[wv][j * DD + h * 8];
          const float4 k1 = *(const float4*)&s_k[wv][j * DD + h * 8 + 4];
          float s = q0.x * k0.x + q0.y * k0.y + q0.z * k0.z + q0.w * k0.w
                  + q1.x * k1.x + q1.y * k1.y + q1.z * k1.z + q1.w * k1.w;
          if ((kmask >> j) & 1u) s = -1e30f;
          sc[j] = s; mx = fmaxf(mx, s);
        }
        float den = 0.f;
#pragma unroll
        for (int j = 0; j < LL; ++j) { sc[j] = __expf(sc[j] - mx); den += sc[j]; }
        const float r = 1.f / den;
        float od[8] = {0.f, 0.f, 0.f, 0.f, 0.f, 0.f, 0.f, 0.f};
#pragma unroll
        for (int j = 0; j < LL; ++j) {
          const float pj = sc[j] * r;
          const float4 v0 = *(const float4*)&s_v[wv][j * DD + h * 8];
          const float4 v1 = *(const float4*)&s_v[wv][j * DD + h * 8 + 4];
          od[0] = fmaf(pj, v0.x, od[0]); od[1] = fmaf(pj, v0.y, od[1]);
          od[2] = fmaf(pj, v0.z, od[2]); od[3] = fmaf(pj, v0.w, od[3]);
          od[4] = fmaf(pj, v1.x, od[4]); od[5] = fmaf(pj, v1.y, od[5]);
          od[6] = fmaf(pj, v1.z, od[6]); od[7] = fmaf(pj, v1.w, od[7]);
        }
#pragma unroll
        for (int d = 0; d < 8; ++d) s_o[wv][i * DD + h * 8 + d] = (_Float16)od[d];
      }
      wave_lds_sync();

      // out-projection (2 WMMAs) fused with masked mean-pool over query rows
      const v16h a_o = load_a_frag(s_o[wv], lane);
      const float rcnt = 1.f / (float)__popc(qval);
      float rmask[8];
#pragma unroll
      for (int j = 0; j < 8; ++j)
        rmask[j] = ((qval >> (mrow + j)) & 1u) ? 1.f : 0.f;
#pragma unroll
      for (int nt = 0; nt < 2; ++nt) {
        v8f c = {}; c = wmma_f16(a_o, load_b_frag(w_out, nt * 16, lane), c);
        const float bo = b_out[nt * 16 + ncol];
        float partial = 0.f;
#pragma unroll
        for (int j = 0; j < 8; ++j)
          partial = fmaf(rmask[j], c[j] + bo, partial);
        s_part[wv][nt * 32 + lane] = partial;
      }
      wave_lds_sync();
      {
        const int t = lane >> 4, c16 = lane & 15;
        const float vsum = s_part[wv][t * 32 + c16] + s_part[wv][t * 32 + c16 + 16];
        s_vec[wv][dir * DD + lane] = vsum * rcnt;   // lane == t*16+c16 == channel
      }
      wave_lds_sync();
    }

    // ---- 4. gate + mix + LayerNorm (lane n owns channel n) ----------------
    {
      float acc = s_gate_b[lane];
      const float* gwp = &s_gate_w[lane * 2 * DD];
#pragma unroll
      for (int k = 0; k < 2 * DD; ++k) acc = fmaf(s_vec[wv][k], gwp[k], acc);
      const float g = 1.f / (1.f + __expf(-acc));
      const float m = g * s_vec[wv][lane] + (1.f - g) * s_vec[wv][DD + lane];

      float s = m;
#pragma unroll
      for (int off = 16; off; off >>= 1) s += __shfl_xor(s, off, 32);
      const float mu = s * (1.f / 32.f);
      const float dv = m - mu;
      float s2 = dv * dv;
#pragma unroll
      for (int off = 16; off; off >>= 1) s2 += __shfl_xor(s2, off, 32);
      const float var = s2 * (1.f / 32.f);

      out[(size_t)b * DD + lane] =
          dv * rsqrtf(var + 1e-5f) * s_ln_w[lane] + s_ln_b[lane];
    }
    wave_lds_sync();   // protect per-wave tiles before next batch overwrites
  }
}

extern "C" void kernel_launch(void* const* d_in, const int* in_sizes, int n_in,
                              void* d_out, int out_size, void* d_ws, size_t ws_size,
                              hipStream_t stream) {
  const int* home_p = (const int*)d_in[0];
  const int* away_p = (const int*)d_in[1];
  const int* h_pos  = (const int*)d_in[2];
  const int* a_pos  = (const int*)d_in[3];
  const float* emb     = (const float*)d_in[4];
  const float* pos_emb = (const float*)d_in[5];
  const float* h2a_in_w  = (const float*)d_in[6];
  const float* h2a_in_b  = (const float*)d_in[7];
  const float* h2a_out_w = (const float*)d_in[8];
  const float* h2a_out_b = (const float*)d_in[9];
  const float* a2h_in_w  = (const float*)d_in[10];
  const float* a2h_in_b  = (const float*)d_in[11];
  const float* a2h_out_w = (const float*)d_in[12];
  const float* a2h_out_b = (const float*)d_in[13];
  const float* gate_w = (const float*)d_in[14];
  const float* gate_b = (const float*)d_in[15];
  const float* ln_w   = (const float*)d_in[16];
  const float* ln_b   = (const float*)d_in[17];
  float* out = (float*)d_out;

  const int nbatch = in_sizes[0] / LL;   // 32768
  dim3 grid(1024), block(NW * 32);       // 4096 waves, 8 batches each
  hipLaunchKernelGGL(crossteam_kernel, grid, block, 0, stream,
                     home_p, away_p, h_pos, a_pos, emb, pos_emb,
                     h2a_in_w, h2a_in_b, h2a_out_w, h2a_out_b,
                     a2h_in_w, a2h_in_b, a2h_out_w, a2h_out_b,
                     gate_w, gate_b, ln_w, ln_b, out, nbatch);
}
